// Tacotron2_37142877176577
// MI455X (gfx1250) — compile-verified, matrix operand order checked by reading, not run on hardware
//
#include <hip/hip_runtime.h>
#include <stdint.h>

// ---------------------------------------------------------------------------
// Tacotron2 decoder for MI455X (gfx1250, wave32).
//  - bf16 WMMA (v_wmma_f32_16x16x32_bf16) for all batchable GEMMs
//  - async global->LDS panel staging (global_load_async_to_lds_b128),
//    double-buffered with in-order ASYNCcnt fencing (s_wait_asynccnt 4)
//    so panel DMA overlaps WMMA
//  - VALU bf16 mat-vec for the sequential LSTM chain (bandwidth-bound, N=1)
//  - global_prefetch_b8 via __builtin_prefetch
// ---------------------------------------------------------------------------

#define BB 16
#define TENC 512
#define TDEC 128
#define EE 512
#define HH 1024
#define G4 4096
#define PRE 256

typedef __attribute__((ext_vector_type(16))) __bf16 v16bf;
typedef __attribute__((ext_vector_type(8)))  float  v8f;

__device__ __forceinline__ unsigned short f2bf(float f) {
  union { float f; unsigned u; } x; x.f = f;
  unsigned u = x.u;
  unsigned r = (u + 0x7FFFu + ((u >> 16) & 1u)) >> 16;
  return (unsigned short)r;
}
__device__ __forceinline__ float bf2f(unsigned short b) {
  union { unsigned u; float f; } x; x.u = ((unsigned)b) << 16; return x.f;
}
__device__ __forceinline__ float sigf(float x) { return 1.0f / (1.0f + __expf(-x)); }

__device__ __forceinline__ unsigned hash3(unsigned a, unsigned b, unsigned c) {
  unsigned h = a * 0x9E3779B1u ^ b * 0x85EBCA77u ^ c * 0xC2B2AE3Du;
  h ^= h >> 15; h *= 0x2C1B3C6Du; h ^= h >> 12; h *= 0x297A2D39u; h ^= h >> 15;
  return h;
}

// ------------------------------- utility kernels ---------------------------

__global__ void k_cast_bf16(const float* __restrict__ src,
                            unsigned short* __restrict__ dst, int n) {
  for (int i = blockIdx.x * blockDim.x + threadIdx.x; i < n;
       i += gridDim.x * blockDim.x)
    dst[i] = f2bf(src[i]);
}

__global__ void k_addvec(const float* __restrict__ a, const float* __restrict__ b,
                         float* __restrict__ d, int n) {
  int i = blockIdx.x * blockDim.x + threadIdx.x;
  if (i < n) d[i] = a[i] + b[i];
}

// ------------------------------- prenet ------------------------------------
__global__ void k_prenet(const float* __restrict__ mels,
                         const unsigned short* __restrict__ Wp1,
                         const unsigned short* __restrict__ Wp2,
                         unsigned short* __restrict__ P) {
  int tb = blockIdx.x;           // t*16 + b
  int t = tb >> 4, b = tb & 15;
  int tid = threadIdx.x;
  __shared__ float x[80];
  __shared__ float p1[PRE];
  if (tid < 80)
    x[tid] = (t == 0) ? 0.0f : mels[(b * 80 + tid) * TDEC + (t - 1)];
  __syncthreads();
  {
    float acc = 0.0f;
    const unsigned short* w = Wp1 + tid * 80;
    for (int k = 0; k < 80; k++) acc += bf2f(w[k]) * x[k];
    acc = fmaxf(acc, 0.0f);
    p1[tid] = (hash3(tb, tid, 0) & 1u) ? 2.0f * acc : 0.0f;
  }
  __syncthreads();
  {
    float acc = 0.0f;
    const unsigned short* w = Wp2 + tid * PRE;
    for (int k = 0; k < PRE; k++) acc += bf2f(w[k]) * p1[k];
    acc = fmaxf(acc, 0.0f);
    float v = (hash3(tb, tid, 1) & 1u) ? 2.0f * acc : 0.0f;
    P[tb * PRE + tid] = f2bf(v);
  }
}

// ------------------------- WMMA tile GEMM (bf16 -> f32) --------------------
// D[m,n] = sum_k A[m,k] * W[n,k]  (+bias[n]) (+Cin[m,n])
// One wave per 16x16 tile. A/B 16x32 bf16 panels are DMA'd into LDS with
// global_load_async_to_lds_b128. Double-buffered: after issuing the 4-op
// group for panel i+1, s_wait_asynccnt 4 guarantees panel i landed (async
// loads complete in order) while panel i+1 stays in flight behind the WMMA.
__global__ __launch_bounds__(32)
void k_gemm_wmma(const unsigned short* __restrict__ A, int lda,
                 const unsigned short* __restrict__ W, int ldw,
                 const float* __restrict__ bias,
                 const float* __restrict__ Cin,
                 float* __restrict__ C, int ldc,
                 int N16, int K) {
  __shared__ __attribute__((aligned(16))) unsigned char smem[4096];
  // layout: A panels at [0,1024) buf0, [1024,2048) buf1;
  //         B panels at [2048,3072) buf0, [3072,4096) buf1.
  int tile = blockIdx.x;
  int m0 = (tile / N16) * 16;
  int n0 = (tile % N16) * 16;
  int lane = threadIdx.x & 31;
  int half = lane >> 4;          // 0: lanes 0-15, 1: lanes 16-31
  int l16 = lane & 15;

  unsigned lds_base = (unsigned)(uintptr_t)(&smem[0]);   // LDS aperture: low 32b
  // per-lane panel staging addresses: 16 rows x 64B; lane covers idx, idx+32
  int r0 = lane >> 2, s0 = lane & 3;       // idx0 = lane      -> rows 0..7
  int r1 = 8 + r0;                          // idx1 = lane + 32 -> rows 8..15
  unsigned long long agA0 = (unsigned long long)(uintptr_t)A +
      ((unsigned long long)(m0 + r0) * lda) * 2ull + (unsigned)s0 * 16u;
  unsigned long long agA1 = (unsigned long long)(uintptr_t)A +
      ((unsigned long long)(m0 + r1) * lda) * 2ull + (unsigned)s0 * 16u;
  unsigned long long agB0 = (unsigned long long)(uintptr_t)W +
      ((unsigned long long)(n0 + r0) * ldw) * 2ull + (unsigned)s0 * 16u;
  unsigned long long agB1 = (unsigned long long)(uintptr_t)W +
      ((unsigned long long)(n0 + r1) * ldw) * 2ull + (unsigned)s0 * 16u;
  unsigned ldsLane = lane * 16u;

#define STAGE_PANEL(buf)                                                      \
  do {                                                                        \
    unsigned dA0 = lds_base + (unsigned)(buf)*1024u + ldsLane;                \
    unsigned dA1 = dA0 + 512u;                                                \
    unsigned dB0 = lds_base + 2048u + (unsigned)(buf)*1024u + ldsLane;        \
    unsigned dB1 = dB0 + 512u;                                                \
    asm volatile("global_load_async_to_lds_b128 %0, %1, off"                  \
                 :: "v"(dA0), "v"(agA0) : "memory");                          \
    asm volatile("global_load_async_to_lds_b128 %0, %1, off"                  \
                 :: "v"(dA1), "v"(agA1) : "memory");                          \
    asm volatile("global_load_async_to_lds_b128 %0, %1, off"                  \
                 :: "v"(dB0), "v"(agB0) : "memory");                          \
    asm volatile("global_load_async_to_lds_b128 %0, %1, off"                  \
                 :: "v"(dB1), "v"(agB1) : "memory");                          \
    agA0 += 64ull; agA1 += 64ull; agB0 += 64ull; agB1 += 64ull;               \
  } while (0)

  STAGE_PANEL(0);

  v8f acc = {};
  for (int kk = 0; kk < K; kk += 32) {
    int cur = (kk >> 5) & 1;
    if (kk + 32 < K) {
      STAGE_PANEL(1 - cur);                       // async DMA of next panel
      __builtin_prefetch(&A[(long)(m0 + l16) * lda + kk + 64], 0, 1);
      __builtin_prefetch(&W[(long)(n0 + l16) * ldw + kk + 64], 0, 1);
      // panel `cur` is the older group: in-order completion => <=4 pending
      // means it has fully landed; panel 1-cur stays in flight behind wmma.
      asm volatile("s_wait_asynccnt 0x4" ::: "memory");
    } else {
      asm volatile("s_wait_asynccnt 0x0" ::: "memory");   // last panel
    }
    const unsigned* As32 = (const unsigned*)(smem + cur * 1024);
    const unsigned* Bs32 = (const unsigned*)(smem + 2048 + cur * 1024);
    union { v16bf v; unsigned u[8]; } af, bfv;
#pragma unroll
    for (int vi = 0; vi < 8; vi++) {
      // A 16x32 bf16 frag: VGPR vi holds K = (vi<4?0:16)+(lane<16?0:8)+2*(vi&3)
      int k0 = ((vi & 4) << 2) + (half << 3) + ((vi & 3) << 1);
      af.u[vi]  = As32[l16 * 16 + (k0 >> 1)];
      bfv.u[vi] = Bs32[l16 * 16 + (k0 >> 1)];
    }
    acc = __builtin_amdgcn_wmma_f32_16x16x32_bf16(
        false, af.v, false, bfv.v, (short)0, acc, false, false);
  }
#undef STAGE_PANEL

  float bv = bias ? bias[n0 + l16] : 0.0f;
  if (Cin) {
#pragma unroll
    for (int r = 0; r < 8; r++) {
      long idx = (long)(m0 + r + 8 * half) * ldc + n0 + l16;
      C[idx] = acc[r] + bv + Cin[idx];
    }
  } else {
#pragma unroll
    for (int r = 0; r < 8; r++) {
      long idx = (long)(m0 + r + 8 * half) * ldc + n0 + l16;
      C[idx] = acc[r] + bv;
    }
  }
}

// ------------------- sequential 2-layer LSTM chain (per step) --------------
__device__ __forceinline__ float dot_bf(const unsigned short* __restrict__ w,
                                        const float* __restrict__ hs) {
  float s = 0.0f;
  for (int k = 0; k < HH; k += 8) {
    uint4 wv = *(const uint4*)(w + k);
    s += bf2f((unsigned short)(wv.x & 0xFFFFu)) * hs[k + 0];
    s += bf2f((unsigned short)(wv.x >> 16))     * hs[k + 1];
    s += bf2f((unsigned short)(wv.y & 0xFFFFu)) * hs[k + 2];
    s += bf2f((unsigned short)(wv.y >> 16))     * hs[k + 3];
    s += bf2f((unsigned short)(wv.z & 0xFFFFu)) * hs[k + 4];
    s += bf2f((unsigned short)(wv.z >> 16))     * hs[k + 5];
    s += bf2f((unsigned short)(wv.w & 0xFFFFu)) * hs[k + 6];
    s += bf2f((unsigned short)(wv.w >> 16))     * hs[k + 7];
  }
  return s;
}

__global__ __launch_bounds__(1024)
void k_chain(const float* __restrict__ G1,
             const unsigned short* __restrict__ Whh1,
             const unsigned short* __restrict__ Wih2,
             const unsigned short* __restrict__ Whh2,
             const float* __restrict__ bsum2,
             float* __restrict__ Hout,
             unsigned short* __restrict__ Hbf) {
  __shared__ float h1[HH];
  __shared__ float h2[HH];
  int tid = threadIdx.x;
  h1[tid] = 0.0f; h2[tid] = 0.0f;
  float c1 = 0.0f, c2 = 0.0f;
  __syncthreads();
  for (int b = 0; b < BB; b++) {
    float acc[4];
#pragma unroll
    for (int j = 0; j < 4; j++) {          // j: gate i,f,g,o for dim=tid
      int r = tid + j * HH;
      acc[j] = G1[b * G4 + r] + dot_bf(Whh1 + (long)r * HH, h1);
    }
    __syncthreads();
    c1 = sigf(acc[1]) * c1 + sigf(acc[0]) * tanhf(acc[2]);
    float nh1 = sigf(acc[3]) * tanhf(c1);
    h1[tid] = nh1;
    __syncthreads();
#pragma unroll
    for (int j = 0; j < 4; j++) {
      int r = tid + j * HH;
      acc[j] = bsum2[r] + dot_bf(Wih2 + (long)r * HH, h1)
                        + dot_bf(Whh2 + (long)r * HH, h2);
    }
    __syncthreads();
    c2 = sigf(acc[1]) * c2 + sigf(acc[0]) * tanhf(acc[2]);
    float nh2 = sigf(acc[3]) * tanhf(c2);
    h2[tid] = nh2;
    Hout[b * HH + tid] = nh2;
    Hbf[b * HH + tid] = f2bf(nh2);
    __syncthreads();
  }
}

// ------------------------- attention (one block per b) ---------------------
__global__ __launch_bounds__(512)
void k_attn(const float* __restrict__ q, float* __restrict__ cum,
            const float* __restrict__ proc_enc, const float* __restrict__ inputs,
            const float* __restrict__ Wc, const float* __restrict__ bc,
            const float* __restrict__ Wld, const float* __restrict__ bld,
            const float* __restrict__ we, const float* __restrict__ be,
            float* __restrict__ ctx, unsigned short* __restrict__ ctx_bf) {
  int b = blockIdx.x;
  int te = threadIdx.x;
  __shared__ float wld_s[128 * 32];
  __shared__ float wc_s[32 * 31];
  __shared__ float qb[128], we_s[128], bld_s[128], bc_s[32];
  __shared__ float cumpad[TENC + 30];
  __shared__ float aw_s[TENC];
  __shared__ float red[512];
  for (int i = te; i < 128 * 32; i += 512) wld_s[i] = Wld[i];
  for (int i = te; i < 32 * 31;  i += 512) wc_s[i] = Wc[i];
  if (te < 128) { qb[te] = q[b * 128 + te]; we_s[te] = we[te]; bld_s[te] = bld[te]; }
  if (te < 32) bc_s[te] = bc[te];
  cumpad[te + 15] = cum[b * TENC + te];
  if (te < 15) { cumpad[te] = 0.0f; cumpad[te + TENC + 15] = 0.0f; }
  __syncthreads();

  float s[32];
  for (int ch = 0; ch < 32; ch++) {        // location conv, 31-tap, pad 15
    float a = bc_s[ch];
    const float* w = &wc_s[ch * 31];
    for (int k = 0; k < 31; k++) a += w[k] * cumpad[te + k];
    s[ch] = a;
  }
  const float* pe = proc_enc + ((long)(b * TENC) + te) * 128;
  float eacc = 0.0f;
  for (int d = 0; d < 128; d++) {
    float loc = bld_s[d];
    const float* w = &wld_s[d * 32];
    for (int ch = 0; ch < 32; ch++) loc += w[ch] * s[ch];
    eacc += we_s[d] * tanhf(qb[d] + loc + pe[d]);
  }
  eacc += be[0];
  // softmax over encoder axis
  red[te] = eacc; __syncthreads();
  for (int off = 256; off; off >>= 1) {
    if (te < off) red[te] = fmaxf(red[te], red[te + off]);
    __syncthreads();
  }
  float mx = red[0]; __syncthreads();
  float ex = __expf(eacc - mx);
  red[te] = ex; __syncthreads();
  for (int off = 256; off; off >>= 1) {
    if (te < off) red[te] += red[te + off];
    __syncthreads();
  }
  float aw = ex / red[0];
  aw_s[te] = aw;
  cum[b * TENC + te] += aw;
  __syncthreads();
  // context: e = te (0..511)
  float cacc = 0.0f;
  const float* inb = inputs + (long)b * TENC * EE;
  for (int t2 = 0; t2 < TENC; t2++) cacc += aw_s[t2] * inb[(long)t2 * EE + te];
  ctx[b * EE + te] = cacc;
  ctx_bf[b * EE + te] = f2bf(cacc);
}

// ------------------------- frame / stop projection -------------------------
__global__ void k_out(const float* __restrict__ Hm, const float* __restrict__ ctx,
                      const float* __restrict__ Wproj, const float* __restrict__ bproj,
                      const float* __restrict__ Wstop, const float* __restrict__ bstop,
                      float* __restrict__ out, int t) {
  int b = blockIdx.x;
  int o = threadIdx.x;
  if (o > 80) return;
  const float* w = (o < 80) ? (Wproj + o * 1536) : Wstop;
  float acc = (o < 80) ? bproj[o] : bstop[0];
  const float* hb = Hm + b * HH;
  const float* cb = ctx + b * EE;
  for (int k = 0; k < HH; k++) acc += w[k] * hb[k];
  for (int k = 0; k < EE; k++) acc += w[HH + k] * cb[k];
  if (o < 80) out[((long)(b * TDEC) + t) * 80 + o] = acc;
  else        out[(long)BB * TDEC * 80 + b * TDEC + t] = acc;
}

// --------------------------------- host ------------------------------------

extern "C" void kernel_launch(void* const* d_in, const int* in_sizes, int n_in,
                              void* d_out, int out_size, void* d_ws, size_t ws_size,
                              hipStream_t stream) {
  (void)in_sizes; (void)n_in; (void)out_size; (void)ws_size;
  const float* inputs   = (const float*)d_in[0];
  const float* mels     = (const float*)d_in[1];
  const float* W_enc    = (const float*)d_in[2];
  const float* W_q      = (const float*)d_in[3];
  const float* W_lc     = (const float*)d_in[4];
  const float* b_lc     = (const float*)d_in[5];
  const float* W_ld     = (const float*)d_in[6];
  const float* b_ld     = (const float*)d_in[7];
  const float* w_e      = (const float*)d_in[8];
  const float* b_e      = (const float*)d_in[9];
  const float* W_p1     = (const float*)d_in[10];
  const float* W_p2     = (const float*)d_in[11];
  const float* W_ih1    = (const float*)d_in[12];
  const float* W_hh1    = (const float*)d_in[13];
  const float* b_ih1    = (const float*)d_in[14];
  const float* b_hh1    = (const float*)d_in[15];
  const float* W_ih2    = (const float*)d_in[16];
  const float* W_hh2    = (const float*)d_in[17];
  const float* b_ih2    = (const float*)d_in[18];
  const float* b_hh2    = (const float*)d_in[19];
  const float* W_proj   = (const float*)d_in[20];
  const float* b_proj   = (const float*)d_in[21];
  const float* W_stop   = (const float*)d_in[22];
  const float* b_stop   = (const float*)d_in[23];
  float* out = (float*)d_out;

  // ---- workspace carve (bump allocator, 256B aligned) ----
  char* p = (char*)d_ws;
  auto carve = [&](size_t bytes) -> void* {
    void* r = (void*)p;
    p += (bytes + 255) & ~(size_t)255;
    return r;
  };
  unsigned short* inputs_bf = (unsigned short*)carve((size_t)BB * TENC * EE * 2);
  unsigned short* Wenc_bf   = (unsigned short*)carve((size_t)128 * EE * 2);
  unsigned short* Wq_bf     = (unsigned short*)carve((size_t)128 * HH * 2);
  unsigned short* Wih1_bf   = (unsigned short*)carve((size_t)G4 * 768 * 2);
  unsigned short* Whh1_bf   = (unsigned short*)carve((size_t)G4 * HH * 2);
  unsigned short* Wih2_bf   = (unsigned short*)carve((size_t)G4 * HH * 2);
  unsigned short* Whh2_bf   = (unsigned short*)carve((size_t)G4 * HH * 2);
  unsigned short* Wp1_bf    = (unsigned short*)carve((size_t)PRE * 80 * 2);
  unsigned short* Wp2_bf    = (unsigned short*)carve((size_t)PRE * PRE * 2);
  unsigned short* P_bf      = (unsigned short*)carve((size_t)TDEC * BB * PRE * 2);
  float* G1p      = (float*)carve((size_t)TDEC * BB * G4 * 4);
  float* proc_enc = (float*)carve((size_t)BB * TENC * 128 * 4);
  float* bsum1    = (float*)carve((size_t)G4 * 4);
  float* bsum2    = (float*)carve((size_t)G4 * 4);
  float* G1       = (float*)carve((size_t)BB * G4 * 4);
  float* Hbuf     = (float*)carve((size_t)BB * HH * 4);
  unsigned short* H_bf   = (unsigned short*)carve((size_t)BB * HH * 2);
  float* ctx      = (float*)carve((size_t)BB * EE * 4);
  unsigned short* ctx_bf = (unsigned short*)carve((size_t)BB * EE * 2);
  float* qbuf     = (float*)carve((size_t)BB * 128 * 4);
  float* cum      = (float*)carve((size_t)BB * TENC * 4);

  auto cast = [&](const float* s, unsigned short* d, int n) {
    int blocks = (n + 256 * 8 - 1) / (256 * 8);
    if (blocks < 1) blocks = 1;
    k_cast_bf16<<<blocks, 256, 0, stream>>>(s, d, n);
  };
  cast(inputs, inputs_bf, BB * TENC * EE);
  cast(W_enc, Wenc_bf, 128 * EE);
  cast(W_q, Wq_bf, 128 * HH);
  cast(W_ih1, Wih1_bf, G4 * 768);
  cast(W_hh1, Whh1_bf, G4 * HH);
  cast(W_ih2, Wih2_bf, G4 * HH);
  cast(W_hh2, Whh2_bf, G4 * HH);
  cast(W_p1, Wp1_bf, PRE * 80);
  cast(W_p2, Wp2_bf, PRE * PRE);

  k_addvec<<<(G4 + 255) / 256, 256, 0, stream>>>(b_ih1, b_hh1, bsum1, G4);
  k_addvec<<<(G4 + 255) / 256, 256, 0, stream>>>(b_ih2, b_hh2, bsum2, G4);

  hipMemsetAsync(cum, 0, (size_t)BB * TENC * 4, stream);
  hipMemsetAsync(ctx, 0, (size_t)BB * EE * 4, stream);
  hipMemsetAsync(ctx_bf, 0, (size_t)BB * EE * 2, stream);

  // prenet for all steps (teacher-forced inputs are fully known up front)
  k_prenet<<<TDEC * BB, PRE, 0, stream>>>(mels, Wp1_bf, Wp2_bf, P_bf);

  // proc_enc = inputs @ W_enc^T : M=8192, N=128, K=512
  k_gemm_wmma<<<(BB * TENC / 16) * (128 / 16), 32, 0, stream>>>(
      inputs_bf, EE, Wenc_bf, EE, nullptr, nullptr, proc_enc, 128, 128 / 16, EE);

  // G1p[t,b,:] = P[t,b,:] @ W_ih1[:, :256]^T : M=2048, N=4096, K=256
  k_gemm_wmma<<<(TDEC * BB / 16) * (G4 / 16), 32, 0, stream>>>(
      P_bf, PRE, Wih1_bf, 768, nullptr, nullptr, G1p, G4, G4 / 16, PRE);

  for (int t = 0; t < TDEC; t++) {
    // gate1 = G1p[t] + ctx @ W_ih1[:, 256:]^T + (b_ih1 + b_hh1)
    k_gemm_wmma<<<(BB / 16) * (G4 / 16), 32, 0, stream>>>(
        ctx_bf, EE, Wih1_bf + 256, 768, bsum1, G1p + (size_t)t * BB * G4,
        G1, G4, G4 / 16, EE);
    // sequential batch-chained 2-layer LSTM
    k_chain<<<1, 1024, 0, stream>>>(G1, Whh1_bf, Wih2_bf, Whh2_bf, bsum2,
                                    Hbuf, H_bf);
    // q = H @ W_q^T : M=16, N=128, K=1024
    k_gemm_wmma<<<(BB / 16) * (128 / 16), 32, 0, stream>>>(
        H_bf, HH, Wq_bf, HH, nullptr, nullptr, qbuf, 128, 128 / 16, HH);
    // attention: conv + loc-dense + energies + softmax + context + cum update
    k_attn<<<BB, 512, 0, stream>>>(qbuf, cum, proc_enc, inputs, W_lc, b_lc,
                                   W_ld, b_ld, w_e, b_e, ctx, ctx_bf);
    // frame + stop projections
    k_out<<<BB, 96, 0, stream>>>(Hbuf, ctx, W_proj, b_proj, W_stop, b_stop,
                                 out, t);
  }
}
